// SSDFLUIDLayer_4389456577368
// MI455X (gfx1250) — matrix operand order repaired and checked by reading, not compiled, not run on hardware
//
#include <hip/hip_runtime.h>
#include <math.h>

typedef __attribute__((ext_vector_type(16))) __bf16 v16bf;
typedef __attribute__((ext_vector_type(8)))  __bf16 v8bf;
typedef __attribute__((ext_vector_type(8)))  float  v8f;

#define BB 2
#define NN 2048
#define DD 1024
#define EE 2048
#define FF 4096
#define LCH 128   // scan chunk (cumsum pipeline)
#define LNC 16
#define RCH 512   // recurrence chunk
#define RNC 4

// round-to-nearest-even f32 -> bf16
__device__ inline __bf16 f2bfs(float f) {
    unsigned u = __builtin_bit_cast(unsigned, f);
    unsigned r = u + 0x7FFFu + ((u >> 16) & 1u);
    unsigned short h = (unsigned short)(r >> 16);
    return __builtin_bit_cast(__bf16, h);
}

// ---------------------------------------------------------------------------
// bf16-input / f32-accumulate WMMA GEMM.  All operands pre-converted:
//   A: [M,K] bf16 row-major.  B: [N,K] bf16 row-major (i.e. B^T layout).
//   C f32 (optional, WC) ; Cm bf16 row-major mirror ; Ct bf16 transposed mirror.
//   C[M,N] (+)= epi(A @ B^T + bias) + res
// Block = 256 threads (8 waves), block tile 128x128, wave tile 32x64
// (2x4 subtiles of v_wmma_f32_16x16x32_bf16).  K step = 32.  The K-slice
// staging is a pure b128 copy (global->reg->LDS), zero conversion VALU.
// EPI: 0 none, 1 elu(x)+1, 2 exact gelu, 3 causal tril mask (zero when n>m).
// ---------------------------------------------------------------------------
template<bool ACC, int EPI, bool WC>
__global__ __launch_bounds__(256)
void gemm_bf16(const __bf16* __restrict__ Ag, int lda, size_t sA,
               const __bf16* __restrict__ Bg, int ldb, size_t sB,
               float* __restrict__ Cg, int ldc, size_t sC,
               __bf16* __restrict__ Cm, int ldm, size_t sM,
               __bf16* __restrict__ Ct, int ldt, size_t sT,
               const float* __restrict__ bias,
               const float* __restrict__ res, int ldr, size_t sR,
               int Kdim) {
    const int t     = threadIdx.x;
    const int w     = t >> 5;
    const int lane  = t & 31;
    const int lrow  = lane & 15;
    const int khalf = lane >> 4;
    const int m0 = blockIdx.y * 128;
    const int n0 = blockIdx.x * 128;
    const int wm = (w >> 1) * 32;   // 4 wave rows
    const int wn = (w & 1) * 64;    // 2 wave cols

    Ag += (size_t)blockIdx.z * sA;
    Bg += (size_t)blockIdx.z * sB;
    if (WC)  Cg += (size_t)blockIdx.z * sC;
    if (Cm)  Cm += (size_t)blockIdx.z * sM;
    if (Ct)  Ct += (size_t)blockIdx.z * sT;
    if (res) res += (size_t)blockIdx.z * sR;

    // padded rows: 40 halves (80B) -> conflict-free b128 fragment loads
    __shared__ __bf16 As[128 * 40];
    __shared__ __bf16 Bs[128 * 40];

    v8f acc[2][4];
#pragma unroll
    for (int i = 0; i < 2; i++)
#pragma unroll
        for (int j = 0; j < 4; j++)
#pragma unroll
            for (int r = 0; r < 8; r++) acc[i][j][r] = 0.0f;

    v8bf va[2], vb[2];   // 8-half (16B) prefetch registers
    auto loadT = [&](int kk) {
#pragma unroll
        for (int q = 0; q < 2; q++) {
            int cid = q * 256 + t;           // 512 chunks of 8 halves
            int r = cid >> 2, kc = (cid & 3) * 8;
            va[q] = *(const v8bf*)&Ag[(size_t)(m0 + r) * lda + kk + kc];
            vb[q] = *(const v8bf*)&Bg[(size_t)(n0 + r) * ldb + kk + kc];
        }
    };
    auto storeT = [&]() {
#pragma unroll
        for (int q = 0; q < 2; q++) {
            int cid = q * 256 + t;
            int r = cid >> 2, kc = (cid & 3) * 8;
            *(v8bf*)&As[r * 40 + kc] = va[q];
            *(v8bf*)&Bs[r * 40 + kc] = vb[q];
        }
    };

    loadT(0);
    storeT();

    for (int k0 = 0; k0 < Kdim; k0 += 32) {
        __syncthreads();
        const bool notlast = (k0 + 32 < Kdim);
        if (notlast) loadT(k0 + 32);

        v16bf afr[2], bfr[4];
#pragma unroll
        for (int i = 0; i < 2; i++) {
            // A fragment: lane<16 -> K 0..7 & 16..23 ; lane>=16 -> K 8..15 & 24..31
            const __bf16* ap = &As[(wm + i * 16 + lrow) * 40 + khalf * 8];
            v8bf alo = *(const v8bf*)ap;
            v8bf ahi = *(const v8bf*)(ap + 16);
#pragma unroll
            for (int q = 0; q < 8; q++) { afr[i][q] = alo[q]; afr[i][8 + q] = ahi[q]; }
        }
#pragma unroll
        for (int j = 0; j < 4; j++) {
            // B fragment: lane<16 -> K 0..7 & 16..23 ; lane>=16 -> K 8..15 & 24..31
            const __bf16* bp = &Bs[(wn + j * 16 + lrow) * 40 + khalf * 8];
            v8bf blo = *(const v8bf*)bp;
            v8bf bhi = *(const v8bf*)(bp + 16);
#pragma unroll
            for (int q = 0; q < 8; q++) { bfr[j][q] = blo[q]; bfr[j][8 + q] = bhi[q]; }
        }
#pragma unroll
        for (int i = 0; i < 2; i++)
#pragma unroll
            for (int j = 0; j < 4; j++)
                acc[i][j] = __builtin_amdgcn_wmma_f32_16x16x32_bf16(
                    false, afr[i], false, bfr[j], (short)0, acc[i][j],
                    false, false);
        __syncthreads();
        if (notlast) storeT();
    }

    // Epilogue. C/D layout: VGPR r: lanes 0-15 -> (M=r, N=lane),
    // lanes 16-31 -> (M=r+8, N=lane-16).
#pragma unroll
    for (int i = 0; i < 2; i++)
#pragma unroll
        for (int j = 0; j < 4; j++)
#pragma unroll
            for (int r = 0; r < 8; r++) {
                int m = m0 + wm + i * 16 + r + khalf * 8;
                int n = n0 + wn + j * 16 + lrow;
                float v = acc[i][j][r];
                if (ACC) v += Cg[(size_t)m * ldc + n];
                if (bias) v += bias[n];
                if (EPI == 1) v = (v > 0.0f) ? (v + 1.0f) : expf(v);
                else if (EPI == 2) v = 0.5f * v * (1.0f + erff(v * 0.70710678118654752f));
                else if (EPI == 3) { if (n > m) v = 0.0f; }
                if (res) v += res[(size_t)m * ldr + n];
                if (WC)  Cg[(size_t)m * ldc + n] = v;
                if (Cm)  Cm[(size_t)m * ldm + n] = f2bfs(v);
                if (Ct)  Ct[(size_t)n * ldt + m] = f2bfs(v);
            }
}

// ---------------------------------------------------------------------------
// Tiled transpose + f32->bf16 convert: src [R][C] f32 -> dst [C][R] bf16.
// ---------------------------------------------------------------------------
__global__ __launch_bounds__(256)
void tcvt_kernel(const float* __restrict__ src, __bf16* __restrict__ dst,
                 int R, int C) {
    __shared__ float tile[32][33];
    int c0 = blockIdx.x * 32, r0 = blockIdx.y * 32;
    int tx = threadIdx.x & 31, ty = threadIdx.x >> 5;  // 32x8
#pragma unroll
    for (int i = 0; i < 32; i += 8)
        tile[ty + i][tx] = src[(size_t)(r0 + ty + i) * C + c0 + tx];
    __syncthreads();
#pragma unroll
    for (int i = 0; i < 32; i += 8)
        dst[(size_t)(c0 + ty + i) * R + r0 + tx] = f2bfs(tile[tx][ty + i]);
}

// ---------------------------------------------------------------------------
// LayerNorm: one block per row; bf16 output (feeds GEMM-A only).
// ---------------------------------------------------------------------------
__global__ __launch_bounds__(256)
void ln_kernel(const float* __restrict__ x, const float* __restrict__ g,
               const float* __restrict__ b, __bf16* __restrict__ y, int Dd) {
    size_t row = blockIdx.x;
    const float* xr = x + row * Dd;
    __bf16* yr = y + row * Dd;
    float s = 0.0f, s2 = 0.0f;
    for (int i = threadIdx.x; i < Dd; i += 256) {
        float v = xr[i]; s += v; s2 += v * v;
    }
    __shared__ float sh1[256], sh2[256];
    sh1[threadIdx.x] = s; sh2[threadIdx.x] = s2;
    __syncthreads();
    for (int o = 128; o; o >>= 1) {
        if (threadIdx.x < o) { sh1[threadIdx.x] += sh1[threadIdx.x + o];
                               sh2[threadIdx.x] += sh2[threadIdx.x + o]; }
        __syncthreads();
    }
    float mu  = sh1[0] / Dd;
    float var = sh2[0] / Dd - mu * mu;
    float inv = rsqrtf(var + 1e-5f);
    for (int i = threadIdx.x; i < Dd; i += 256)
        yr[i] = f2bfs((xr[i] - mu) * inv * g[i] + b[i]);
}

// --- denominator pipeline: Z = cumsum(K, axis=seq); den = <Q, Z> + 1e-6 ----
__global__ __launch_bounds__(256)
void chunk_sum_kernel(const __bf16* __restrict__ K, float* __restrict__ cs) {
    int e = blockIdx.x * 256 + threadIdx.x;
    int c = blockIdx.y, b = blockIdx.z;
    const __bf16* p = K + ((size_t)b * NN + (size_t)c * LCH) * EE + e;
    float s = 0.0f;
    for (int i = 0; i < LCH; i++) s += (float)p[(size_t)i * EE];
    cs[((size_t)b * LNC + c) * EE + e] = s;
}

__global__ __launch_bounds__(256)
void chunk_prefix_kernel(float* __restrict__ cs) {
    int e = blockIdx.x * 256 + threadIdx.x;
    int b = blockIdx.z;
    float run = 0.0f;
    for (int c = 0; c < LNC; c++) {
        size_t idx = ((size_t)b * LNC + c) * EE + e;
        float v = cs[idx];
        cs[idx] = run;   // exclusive prefix
        run += v;
    }
}

__global__ __launch_bounds__(256)
void zcum_kernel(const __bf16* __restrict__ K, const float* __restrict__ cs,
                 float* __restrict__ Z) {
    int e = blockIdx.x * 256 + threadIdx.x;
    int c = blockIdx.y, b = blockIdx.z;
    float z = cs[((size_t)b * LNC + c) * EE + e];
    const __bf16* kp = K + ((size_t)b * NN + (size_t)c * LCH) * EE + e;
    float*        zp = Z + ((size_t)b * NN + (size_t)c * LCH) * EE + e;
    for (int i = 0; i < LCH; i++) {
        z += (float)kp[(size_t)i * EE];
        zp[(size_t)i * EE] = z;
    }
}

__global__ __launch_bounds__(256)
void den_kernel(const __bf16* __restrict__ Q, const float* __restrict__ Z,
                float* __restrict__ den) {
    size_t row = blockIdx.x;
    const __bf16* q = Q + row * EE;
    const float*  z = Z + row * EE;
    float s = 0.0f;
    for (int i = threadIdx.x; i < EE; i += 256) s += (float)q[i] * z[i];
    __shared__ float sh[256];
    sh[threadIdx.x] = s; __syncthreads();
    for (int o = 128; o; o >>= 1) {
        if (threadIdx.x < o) sh[threadIdx.x] += sh[threadIdx.x + o];
        __syncthreads();
    }
    if (threadIdx.x == 0) den[row] = sh[0] + 1e-6f;
}

// num f32 -> numbf bf16, divided by den
__global__ __launch_bounds__(256)
void scale_kernel(const float* __restrict__ num, const float* __restrict__ den,
                  __bf16* __restrict__ numbf) {
    size_t idx = (size_t)blockIdx.x * 256 + threadIdx.x;
    numbf[idx] = f2bfs(num[idx] / den[idx / EE]);
}

// ---------------------------------------------------------------------------
extern "C" void kernel_launch(void* const* d_in, const int* in_sizes, int n_in,
                              void* d_out, int out_size, void* d_ws, size_t ws_size,
                              hipStream_t stream) {
    const float* x    = (const float*)d_in[0];
    const float* q_w  = (const float*)d_in[1];
    const float* q_b  = (const float*)d_in[2];
    const float* k_w  = (const float*)d_in[3];
    const float* k_b  = (const float*)d_in[4];
    const float* v_w  = (const float*)d_in[5];
    const float* v_b  = (const float*)d_in[6];
    const float* o_w  = (const float*)d_in[7];
    const float* o_b  = (const float*)d_in[8];
    const float* ln1g = (const float*)d_in[9];
    const float* ln1b = (const float*)d_in[10];
    const float* ln2g = (const float*)d_in[11];
    const float* ln2b = (const float*)d_in[12];
    const float* w1   = (const float*)d_in[13];
    const float* b1   = (const float*)d_in[14];
    const float* w2   = (const float*)d_in[15];
    const float* b2   = (const float*)d_in[16];
    float* out = (float*)d_out;

    const int    M    = BB * NN;                   // 4096 fused rows
    const size_t BND  = (size_t)M * DD * 4;        // 16 MB
    const size_t BNE  = (size_t)M * EE * 4;        // 32 MB
    const size_t BEE  = (size_t)BB * EE * EE * 4;  // 32 MB

    char* base = (char*)d_ws;
    size_t o = 0;
    auto alloc = [&](size_t bytes) { char* p = base + o; o += (bytes + 255) & ~(size_t)255; return p; };

    float* num   = (float*)alloc(BNE);        // [B,N,E] f32 (also Z)
    float* S     = (float*)alloc(BEE);        // [B,E,E] f32 state master
    float* x1    = (float*)alloc(BND);        // [B*N,D] f32
    float* den   = (float*)alloc((size_t)M * 4);
    float* cs    = (float*)alloc((size_t)BB * LNC * EE * 4);
    __bf16* xnbf = (__bf16*)alloc(BND / 2);   // also xn2bf
    __bf16* Qbf  = (__bf16*)alloc(BNE / 2);   // [B*N][E]   (ffnbf aliases Qbf+Kbf)
    __bf16* Kbf  = (__bf16*)alloc(BNE / 2);   // [B*N][E]
    __bf16* KbfT = (__bf16*)alloc(BNE / 2);   // [E][B*N]   (numbf alias)
    __bf16* VbfT = (__bf16*)alloc(BNE / 2);   // [E][B*N]
    __bf16* SbfT = (__bf16*)alloc(BEE / 2);   // [B][E][E] transposed mirror of S
    __bf16* scbf = (__bf16*)alloc((size_t)BB * RCH * RCH * 2);
    __bf16* wqT  = (__bf16*)alloc((size_t)DD * EE * 2);
    __bf16* wkT  = (__bf16*)alloc((size_t)DD * EE * 2);
    __bf16* wvT  = (__bf16*)alloc((size_t)DD * EE * 2);
    __bf16* woT  = (__bf16*)alloc((size_t)EE * DD * 2);
    __bf16* w1T  = (__bf16*)alloc((size_t)DD * FF * 2);
    __bf16* w2T  = (__bf16*)alloc((size_t)FF * DD * 2);
    __bf16* ffnbf = Qbf;    // [B*N][F]: Q,K bf16 dead after attention
    __bf16* numbf = KbfT;   // [B*N][E]: KbfT dead after last S-update
    __bf16* xn2bf = xnbf;   // xn dead after V projection

    const size_t sQe = (size_t)NN * EE;   // batch stride, fused [B*N][E] buffers
    const size_t sSe = (size_t)EE * EE;
    dim3 blk(256);

    // 0) one-time weight transpose+convert (bf16, [out][in] layout)
    tcvt_kernel<<<dim3(EE/32, DD/32), blk, 0, stream>>>(q_w, wqT, DD, EE);
    tcvt_kernel<<<dim3(EE/32, DD/32), blk, 0, stream>>>(k_w, wkT, DD, EE);
    tcvt_kernel<<<dim3(EE/32, DD/32), blk, 0, stream>>>(v_w, wvT, DD, EE);
    tcvt_kernel<<<dim3(DD/32, EE/32), blk, 0, stream>>>(o_w, woT, EE, DD);
    tcvt_kernel<<<dim3(FF/32, DD/32), blk, 0, stream>>>(w1, w1T, DD, FF);
    tcvt_kernel<<<dim3(DD/32, FF/32), blk, 0, stream>>>(w2, w2T, FF, DD);

    // 1) LN1 -> bf16
    ln_kernel<<<dim3(M), blk, 0, stream>>>(x, ln1g, ln1b, xnbf, DD);

    // 2-4) Q,K,V projections (Q,K: elu+1).  bf16-only outputs in GEMM layouts.
    gemm_bf16<false,1,false><<<dim3(EE/128, M/128, 1), blk, 0, stream>>>(
        xnbf, DD, 0, wqT, DD, 0, nullptr, 0, 0,
        Qbf, EE, 0, nullptr, 0, 0, q_b, nullptr, 0, 0, DD);
    gemm_bf16<false,1,false><<<dim3(EE/128, M/128, 1), blk, 0, stream>>>(
        xnbf, DD, 0, wkT, DD, 0, nullptr, 0, 0,
        Kbf, EE, 0, KbfT, M, 0, k_b, nullptr, 0, 0, DD);
    gemm_bf16<false,0,false><<<dim3(EE/128, M/128, 1), blk, 0, stream>>>(
        xnbf, DD, 0, wvT, DD, 0, nullptr, 0, 0,
        nullptr, 0, 0, VbfT, M, 0, v_b, nullptr, 0, 0, DD);

    // 5) denominator: Z = cumsum(K) (into num buffer), den = <Q,Z> + 1e-6
    chunk_sum_kernel<<<dim3(EE/256, LNC, BB), blk, 0, stream>>>(Kbf, cs);
    chunk_prefix_kernel<<<dim3(EE/256, 1, BB), blk, 0, stream>>>(cs);
    zcum_kernel<<<dim3(EE/256, LNC, BB), blk, 0, stream>>>(Kbf, cs, num);
    den_kernel<<<dim3(M), blk, 0, stream>>>(Qbf, num, den);

    // 6) chunked recurrence (chunk = 512): S and its bf16 mirror start at zero
    hipMemsetAsync(S, 0, BEE, stream);
    hipMemsetAsync(SbfT, 0, BEE / 2, stream);
    for (int c = 0; c < RNC; c++) {
        size_t co  = (size_t)c * RCH * EE;   // row offset in [B*N][E] buffers
        size_t cot = (size_t)c * RCH;        // col offset in [E][B*N] buffers
        // scores = tril(Qc @ Kc^T)              [B,512,512], K=E
        gemm_bf16<false,3,false><<<dim3(RCH/128, RCH/128, BB), blk, 0, stream>>>(
            Qbf + co, EE, sQe, Kbf + co, EE, sQe, nullptr, 0, 0,
            scbf, RCH, (size_t)RCH*RCH, nullptr, 0, 0,
            nullptr, nullptr, 0, 0, EE);
        // num = scores @ Vc                     [B,512,E], K=512
        gemm_bf16<false,0,true><<<dim3(EE/128, RCH/128, BB), blk, 0, stream>>>(
            scbf, RCH, (size_t)RCH*RCH, VbfT + cot, M, NN, num + co, EE, sQe,
            nullptr, 0, 0, nullptr, 0, 0,
            nullptr, nullptr, 0, 0, RCH);
        // num += Qc @ S                         [B,512,E], K=E
        gemm_bf16<true,0,true><<<dim3(EE/128, RCH/128, BB), blk, 0, stream>>>(
            Qbf + co, EE, sQe, SbfT, EE, sSe, num + co, EE, sQe,
            nullptr, 0, 0, nullptr, 0, 0,
            nullptr, nullptr, 0, 0, EE);
        // S += Kc^T @ Vc  (+ refresh bf16 mirror)  [B,E,E], K=512
        gemm_bf16<true,0,true><<<dim3(EE/128, EE/128, BB), blk, 0, stream>>>(
            KbfT + cot, M, NN, VbfT + cot, M, NN, S, EE, sSe,
            nullptr, 0, 0, SbfT, EE, sSe,
            nullptr, nullptr, 0, 0, RCH);
    }

    // 7) numbf = (num / den) in bf16
    scale_kernel<<<dim3((unsigned)((size_t)M * EE / 256)), blk, 0, stream>>>(num, den, numbf);

    // 8) x1 = x + num @ o_w + o_b
    gemm_bf16<false,0,true><<<dim3(DD/128, M/128, 1), blk, 0, stream>>>(
        numbf, EE, 0, woT, EE, 0, x1, DD, 0,
        nullptr, 0, 0, nullptr, 0, 0, o_b, x, DD, 0, EE);

    // 9) LN2 -> bf16
    ln_kernel<<<dim3(M), blk, 0, stream>>>(x1, ln2g, ln2b, xn2bf, DD);

    // 10) ffn = gelu(xn2 @ w1 + b1), bf16 only
    gemm_bf16<false,2,false><<<dim3(FF/128, M/128, 1), blk, 0, stream>>>(
        xn2bf, DD, 0, w1T, DD, 0, nullptr, 0, 0,
        ffnbf, FF, 0, nullptr, 0, 0, b1, nullptr, 0, 0, DD);

    // 11) out = x1 + ffn @ w2 + b2
    gemm_bf16<false,0,true><<<dim3(DD/128, M/128, 1), blk, 0, stream>>>(
        ffnbf, FF, 0, w2T, FF, 0, out, DD, 0,
        nullptr, 0, 0, nullptr, 0, 0, b2, x1, DD, 0, FF);
}
